// ClusterLongformerClassifier_21930103014174
// MI455X (gfx1250) — matrix operand order repaired and checked by reading, not compiled, not run on hardware
//
#include <hip/hip_runtime.h>
#include <hip/hip_bf16.h>
#include <cstdint>

typedef unsigned short u16;
typedef __attribute__((ext_vector_type(16))) __bf16 v16bf;
typedef __attribute__((ext_vector_type(8)))  float  v8f;

// ---- problem constants (from reference) ----
constexpr int B_    = 2;
constexpr int S_SRC = 4093;
constexpr int D_    = 768;
constexpr int H_    = 12;
constexpr int FF_   = 3072;
constexpr int L_    = 2;
constexpr int W_    = 256;
constexpr int NG_   = 3;
constexpr int S_    = S_SRC + NG_;   // 4096
constexpr int NC_   = S_ / W_;       // 16
constexpr int MBS   = B_ * S_;       // 8192
constexpr float QSCALE = 0.125f;     // 1/sqrt(64)

__device__ __forceinline__ u16 f2bf(float f) {
  union { float f; uint32_t u; } x; x.f = f;
  uint32_t u = x.u + 0x7fffu + ((x.u >> 16) & 1u);
  return (u16)(u >> 16);
}
__device__ __forceinline__ float bf2f(u16 h) {
  union { uint32_t u; float f; } x; x.u = ((uint32_t)h) << 16;
  return x.f;
}

// gfx1250 async global->LDS copy (VGLOBAL GLOBAL_LOAD_ASYNC_TO_LDS_B128,
// ASYNCcnt-tracked). ldsOff = wave-relative LDS byte offset (low 32 bits of
// the generic pointer), gaddr = 64-bit global VA.
__device__ __forceinline__ void async_copy_b128(uint32_t ldsOff, uint64_t gaddr) {
  asm volatile("global_load_async_to_lds_b128 %0, %1, off"
               :: "v"(ldsOff), "v"(gaddr) : "memory");
}
__device__ __forceinline__ void wait_asynccnt0() {
  asm volatile("s_wait_asynccnt 0x0" ::: "memory");
}
__device__ __forceinline__ uint32_t lds_off32(const void* p) {
  return (uint32_t)(uintptr_t)p;  // LDS aperture: addr[31:0] = LDS byte offset
}

// ---------------------------------------------------------------------------
// Weight convert + transpose: src f32 [K][N] (row-major) -> dst bf16 [N][K]
// ---------------------------------------------------------------------------
__global__ void __launch_bounds__(256) cvt_transpose_kernel(
    const float* __restrict__ src, u16* __restrict__ dst, int K, int N) {
  int idx = blockIdx.x * 256 + threadIdx.x;
  if (idx < K * N) {
    int n = idx / K;
    int k = idx - n * K;
    dst[idx] = f2bf(src[(size_t)k * N + n]);
  }
}

// ---------------------------------------------------------------------------
// bf16 WMMA GEMM:  C[M,N] = A[M,K] * Wt[N,K]^T + bias
//   A  : bf16 row-major [M,K]
//   Wt : bf16 row-major [N,K]  (weights pre-transposed)
//   outF (f32) and/or outB (bf16), either may be null.
// Block tile 128x128, 8 waves, each wave 64x32 => 8 x v_wmma_f32_16x16x32_bf16
// per K-step of 32. Double-buffered LDS fed by async global->LDS copies.
// M%128==0, N%128==0, K%32==0 guaranteed by problem dims.
// ---------------------------------------------------------------------------
union FragBF16 {
  v16bf v;
  uint4 q[2];
};

__global__ void __launch_bounds__(256) gemm_bf16_kernel(
    const u16* __restrict__ A, const u16* __restrict__ Wt,
    const float* __restrict__ bias,
    float* __restrict__ outF, u16* __restrict__ outB,
    int M, int N, int K) {
  (void)M;
  __shared__ u16 As[2][128 * 40];  // row stride 40 ushorts = 80 B
  __shared__ u16 Bs[2][128 * 40];

  const int tid   = threadIdx.x;
  const int lane  = tid & 31;
  const int wave  = tid >> 5;    // 0..7
  const int wm    = wave & 1;    // M sub-tile: wm*64
  const int wn    = wave >> 1;   // N sub-tile: wn*32
  const int half  = lane >> 4;   // lane half (0/1)
  const int l16   = lane & 15;
  const int mBase = blockIdx.y * 128;
  const int nBase = blockIdx.x * 128;

  // Per-thread staging geometry: 512 b128 chunks per tile, 2 per thread.
  const int r0  = tid >> 2;             // chunk 0 row
  const int c80 = (tid & 3) * 8;        // chunk 0 col (bf16 units)
  const int r1  = (tid + 256) >> 2;     // chunk 1 row
  const int c81 = ((tid + 256) & 3) * 8;

  const u16* gA0 = A  + (size_t)(mBase + r0) * K + c80;
  const u16* gA1 = A  + (size_t)(mBase + r1) * K + c81;
  const u16* gB0 = Wt + (size_t)(nBase + r0) * K + c80;
  const u16* gB1 = Wt + (size_t)(nBase + r1) * K + c81;

  auto stage = [&](int k0, int buf) {
    async_copy_b128(lds_off32(&As[buf][r0 * 40 + c80]), (uint64_t)(uintptr_t)(gA0 + k0));
    async_copy_b128(lds_off32(&Bs[buf][r0 * 40 + c80]), (uint64_t)(uintptr_t)(gB0 + k0));
    async_copy_b128(lds_off32(&As[buf][r1 * 40 + c81]), (uint64_t)(uintptr_t)(gA1 + k0));
    async_copy_b128(lds_off32(&Bs[buf][r1 * 40 + c81]), (uint64_t)(uintptr_t)(gB1 + k0));
  };

  v8f acc[4][2];
  {
    v8f zero;
#pragma unroll
    for (int i = 0; i < 8; ++i) zero[i] = 0.0f;
#pragma unroll
    for (int mt = 0; mt < 4; ++mt)
#pragma unroll
      for (int nt = 0; nt < 2; ++nt) acc[mt][nt] = zero;
  }

  stage(0, 0);
  int buf = 0;
  for (int k0 = 0; k0 < K; k0 += 32) {
    wait_asynccnt0();      // this wave's async copies for `buf` landed in LDS
    __syncthreads();       // ...and everyone else's too

    if (k0 + 32 < K) {
      stage(k0 + 32, buf ^ 1);          // prefetch next tile into other buffer
      if (k0 + 64 < K) {                // warm L2 for the tile after next
        __builtin_prefetch(gA0 + k0 + 64, 0, 1);
        __builtin_prefetch(gB0 + k0 + 64, 0, 1);
      }
    }

    FragBF16 af[4], bf[2];
    // A 16x32 bf16 fragment (ISA 7.12.2): lanes 0-15 hold K0-7 (bytes 0-15)
    // and K16-23 (bytes 32-47); lanes 16-31 hold K8-15 and K24-31.
#pragma unroll
    for (int mt = 0; mt < 4; ++mt) {
      int row = wm * 64 + mt * 16 + l16;
      int off = row * 40 + half * 8;            // ushort units
      af[mt].q[0] = *(const uint4*)&As[buf][off];
      af[mt].q[1] = *(const uint4*)&As[buf][off + 16];
    }
    // B 32x16 bf16 fragment: lane n (0-15) holds col n; lanes 0-15 carry
    // K0-7 and K16-23, lanes 16-31 carry K8-15 and K24-31.
#pragma unroll
    for (int nt = 0; nt < 2; ++nt) {
      int row = wn * 32 + nt * 16 + l16;
      int off = row * 40 + half * 8;            // ushort units
      bf[nt].q[0] = *(const uint4*)&Bs[buf][off];
      bf[nt].q[1] = *(const uint4*)&Bs[buf][off + 16];
    }
#pragma unroll
    for (int mt = 0; mt < 4; ++mt)
#pragma unroll
      for (int nt = 0; nt < 2; ++nt)
        acc[mt][nt] = __builtin_amdgcn_wmma_f32_16x16x32_bf16(
            false, af[mt].v, false, bf[nt].v, (short)0, acc[mt][nt], false, false);
    buf ^= 1;
  }

  // Epilogue: C 16x16 f32 layout: VGPR r -> M = r (+8 for lane half), N = lane&15
#pragma unroll
  for (int mt = 0; mt < 4; ++mt) {
#pragma unroll
    for (int nt = 0; nt < 2; ++nt) {
      int rbase = mBase + wm * 64 + mt * 16 + half * 8;
      int col   = nBase + wn * 32 + nt * 16 + l16;
      float bv  = bias ? bias[col] : 0.0f;
#pragma unroll
      for (int r = 0; r < 8; ++r) {
        float v  = acc[mt][nt][r] + bv;
        size_t o = (size_t)(rbase + r) * N + col;
        if (outF) outF[o] = v;
        if (outB) outB[o] = f2bf(v);
      }
    }
  }
}

// ---------------------------------------------------------------------------
// Embedding gather + LayerNorm. One block per token row (b,s). 256 thr x 3.
// ---------------------------------------------------------------------------
__global__ void __launch_bounds__(256) embed_ln_kernel(
    const int* __restrict__ src, const float* __restrict__ wemb,
    const float* __restrict__ pemb, const float* __restrict__ emb_ln,
    float* __restrict__ outF, u16* __restrict__ outB) {
  int bs = blockIdx.x;
  int b  = bs / S_;
  int s  = bs - b * S_;
  int id = (s < NG_) ? (50261 + s) : src[(size_t)b * S_SRC + (s - NG_)];
  const float* w = wemb + (size_t)id * D_;
  const float* p = pemb + (size_t)s  * D_;
  int tid = threadIdx.x;
  float v[3];
#pragma unroll
  for (int i = 0; i < 3; ++i) {
    int d = tid + i * 256;
    v[i] = w[d] + p[d];
  }
  __shared__ float red[256];
  float sm = v[0] + v[1] + v[2];
  red[tid] = sm; __syncthreads();
  for (int o = 128; o > 0; o >>= 1) { if (tid < o) red[tid] += red[tid + o]; __syncthreads(); }
  float mu = red[0] * (1.0f / D_);
  __syncthreads();
  float vs = 0.0f;
#pragma unroll
  for (int i = 0; i < 3; ++i) { float t = v[i] - mu; vs += t * t; }
  red[tid] = vs; __syncthreads();
  for (int o = 128; o > 0; o >>= 1) { if (tid < o) red[tid] += red[tid + o]; __syncthreads(); }
  float rr = rsqrtf(red[0] * (1.0f / D_) + 1e-5f);
#pragma unroll
  for (int i = 0; i < 3; ++i) {
    int d = tid + i * 256;
    float o = (v[i] - mu) * rr * emb_ln[d] + emb_ln[D_ + d];
    size_t idx = (size_t)bs * D_ + d;
    outF[idx] = o;
    outB[idx] = f2bf(o);
  }
}

// ---------------------------------------------------------------------------
// Residual add + LayerNorm: out = LN(a + r) * scale + bias
// ---------------------------------------------------------------------------
__global__ void __launch_bounds__(256) resln_kernel(
    const float* __restrict__ a, const float* __restrict__ r,
    const float* __restrict__ scale, const float* __restrict__ bias,
    float* __restrict__ outF, u16* __restrict__ outB) {
  size_t row = blockIdx.x;
  int tid = threadIdx.x;
  const float* ap = a + row * D_;
  const float* rp = r + row * D_;
  float v[3];
#pragma unroll
  for (int i = 0; i < 3; ++i) {
    int d = tid + i * 256;
    v[i] = ap[d] + rp[d];
  }
  __shared__ float red[256];
  float sm = v[0] + v[1] + v[2];
  red[tid] = sm; __syncthreads();
  for (int o = 128; o > 0; o >>= 1) { if (tid < o) red[tid] += red[tid + o]; __syncthreads(); }
  float mu = red[0] * (1.0f / D_);
  __syncthreads();
  float vs = 0.0f;
#pragma unroll
  for (int i = 0; i < 3; ++i) { float t = v[i] - mu; vs += t * t; }
  red[tid] = vs; __syncthreads();
  for (int o = 128; o > 0; o >>= 1) { if (tid < o) red[tid] += red[tid + o]; __syncthreads(); }
  float rr = rsqrtf(red[0] * (1.0f / D_) + 1e-5f);
#pragma unroll
  for (int i = 0; i < 3; ++i) {
    int d = tid + i * 256;
    float o = (v[i] - mu) * rr * scale[d] + bias[d];
    size_t idx = row * D_ + d;
    outF[idx] = o;
    outB[idx] = f2bf(o);
  }
}

// ---------------------------------------------------------------------------
// Exact GELU, f32 in -> bf16 out.
// ---------------------------------------------------------------------------
__global__ void __launch_bounds__(256) gelu_kernel(
    const float* __restrict__ h, u16* __restrict__ hb, int n) {
  int i0 = blockIdx.x * 1024 + threadIdx.x;
#pragma unroll
  for (int k = 0; k < 4; ++k) {
    int idx = i0 + k * 256;
    if (idx < n) {
      float x = h[idx];
      float g = 0.5f * x * (1.0f + erff(x * 0.70710678f));
      hb[idx] = f2bf(g);
    }
  }
}

// ---------------------------------------------------------------------------
// Local banded attention (Longformer window). Block = one (b,h,chunk) of
// W_=256 queries, one thread per query; online softmax over NG global keys
// plus 12 LDS-staged tiles of 64 local keys.
// ---------------------------------------------------------------------------
__global__ void __launch_bounds__(256) local_attn_kernel(
    const u16* __restrict__ qb, const u16* __restrict__ kb,
    const u16* __restrict__ vb, const unsigned char* __restrict__ mask_src,
    u16* __restrict__ outb) {
  const int c    = blockIdx.x;
  const int h    = blockIdx.y;
  const int b    = blockIdx.z;
  const int qpos = threadIdx.x;
  const int s    = c * W_ + qpos;

  __shared__ u16 Kt[64 * 64];
  __shared__ u16 Vt[64 * 64];
  __shared__ int vf[64];

  float qv[64];
  {
    const u16* qp = qb + ((size_t)(b * S_ + s)) * D_ + h * 64;
#pragma unroll
    for (int d2 = 0; d2 < 32; ++d2) {
      unsigned int u = *(const unsigned int*)(qp + 2 * d2);
      qv[2 * d2]     = bf2f((u16)u) * QSCALE;
      qv[2 * d2 + 1] = bf2f((u16)(u >> 16)) * QSCALE;
    }
  }
  float m = -3.0e38f, l = 0.0f, acc[64];
#pragma unroll
  for (int d = 0; d < 64; ++d) acc[d] = 0.0f;

  // NG global keys (positions 0..NG-1), never masked in s_glb
  for (int g = 0; g < NG_; ++g) {
    const u16* kp = kb + ((size_t)(b * S_ + g)) * D_ + h * 64;
    const u16* vp = vb + ((size_t)(b * S_ + g)) * D_ + h * 64;
    float sd = 0.0f;
#pragma unroll
    for (int d2 = 0; d2 < 32; ++d2) {
      unsigned int u = *(const unsigned int*)(kp + 2 * d2);
      sd += qv[2 * d2] * bf2f((u16)u) + qv[2 * d2 + 1] * bf2f((u16)(u >> 16));
    }
    float nm = fmaxf(m, sd);
    float sc = __expf(m - nm);
    float ex = __expf(sd - nm);
    l = l * sc + ex;
#pragma unroll
    for (int d2 = 0; d2 < 32; ++d2) {
      unsigned int u = *(const unsigned int*)(vp + 2 * d2);
      acc[2 * d2]     = acc[2 * d2]     * sc + ex * bf2f((u16)u);
      acc[2 * d2 + 1] = acc[2 * d2 + 1] * sc + ex * bf2f((u16)(u >> 16));
    }
    m = nm;
  }

  // 12 tiles of 64 keys covering the 3W local window
  for (int t = 0; t < 3 * W_ / 64; ++t) {
    __syncthreads();
#pragma unroll
    for (int i = 0; i < 2; ++i) {
      int cid = threadIdx.x + i * 256;
      int j   = cid >> 3;
      int c8  = (cid & 7) * 8;
      int kpj = c * W_ - W_ + t * 64 + j;
      bool ok = (kpj >= NG_) && (kpj < S_);
      if (ok) ok = mask_src[(size_t)b * S_SRC + (kpj - NG_)] != 0;
      uint4 kz = make_uint4(0, 0, 0, 0), vz = make_uint4(0, 0, 0, 0);
      if (ok) {
        kz = *(const uint4*)(kb + ((size_t)(b * S_ + kpj)) * D_ + h * 64 + c8);
        vz = *(const uint4*)(vb + ((size_t)(b * S_ + kpj)) * D_ + h * 64 + c8);
      }
      *(uint4*)&Kt[j * 64 + c8] = kz;
      *(uint4*)&Vt[j * 64 + c8] = vz;
      if ((cid & 7) == 0) vf[j] = ok ? 1 : 0;
    }
    __syncthreads();
    for (int j = 0; j < 64; ++j) {
      int rel = (t * 64 + j) - W_ - qpos;
      if (rel > W_ || rel < -W_ || !vf[j]) continue;
      float sd = 0.0f;
#pragma unroll
      for (int d2 = 0; d2 < 32; ++d2) {
        unsigned int u = *(const unsigned int*)&Kt[j * 64 + 2 * d2];
        sd += qv[2 * d2] * bf2f((u16)u) + qv[2 * d2 + 1] * bf2f((u16)(u >> 16));
      }
      float nm = fmaxf(m, sd);
      float sc = __expf(m - nm);
      float ex = __expf(sd - nm);
      l = l * sc + ex;
#pragma unroll
      for (int d2 = 0; d2 < 32; ++d2) {
        unsigned int u = *(const unsigned int*)&Vt[j * 64 + 2 * d2];
        acc[2 * d2]     = acc[2 * d2]     * sc + ex * bf2f((u16)u);
        acc[2 * d2 + 1] = acc[2 * d2 + 1] * sc + ex * bf2f((u16)(u >> 16));
      }
      m = nm;
    }
  }

  float inv = 1.0f / l;
  u16* op = outb + ((size_t)(b * S_ + s)) * D_ + h * 64;
#pragma unroll
  for (int d2 = 0; d2 < 32; ++d2) {
    unsigned int u = (unsigned int)f2bf(acc[2 * d2] * inv) |
                     ((unsigned int)f2bf(acc[2 * d2 + 1] * inv) << 16);
    *(unsigned int*)(op + 2 * d2) = u;
  }
}

// ---------------------------------------------------------------------------
// Global-token attention: NG query tokens attend over all S keys.
// One block per batch; thread t handles (g = t%NG, h = t/NG). Overwrites
// rows [0, NG) of the local-attention output.
// ---------------------------------------------------------------------------
__global__ void __launch_bounds__(64) global_attn_kernel(
    const u16* __restrict__ qgb, const u16* __restrict__ kgb,
    const u16* __restrict__ vgb, const unsigned char* __restrict__ mask_src,
    u16* __restrict__ outb) {
  int b = blockIdx.x;
  int t = threadIdx.x;
  if (t >= NG_ * H_) return;
  int g = t % NG_;
  int h = t / NG_;

  float qv[64];
  {
    const u16* qp = qgb + ((size_t)(b * S_ + g)) * D_ + h * 64;
#pragma unroll
    for (int d2 = 0; d2 < 32; ++d2) {
      unsigned int u = *(const unsigned int*)(qp + 2 * d2);
      qv[2 * d2]     = bf2f((u16)u) * QSCALE;
      qv[2 * d2 + 1] = bf2f((u16)(u >> 16)) * QSCALE;
    }
  }
  float m = -3.0e38f, l = 0.0f, acc[64];
#pragma unroll
  for (int d = 0; d < 64; ++d) acc[d] = 0.0f;

  for (int s2 = 0; s2 < S_; ++s2) {
    bool ok = (s2 < NG_) || (mask_src[(size_t)b * S_SRC + (s2 - NG_)] != 0);
    if (!ok) continue;
    const u16* kp = kgb + ((size_t)(b * S_ + s2)) * D_ + h * 64;
    float sd = 0.0f;
#pragma unroll
    for (int d2 = 0; d2 < 32; ++d2) {
      unsigned int u = *(const unsigned int*)(kp + 2 * d2);
      sd += qv[2 * d2] * bf2f((u16)u) + qv[2 * d2 + 1] * bf2f((u16)(u >> 16));
    }
    float nm = fmaxf(m, sd);
    float sc = __expf(m - nm);
    float ex = __expf(sd - nm);
    l = l * sc + ex;
    const u16* vp = vgb + ((size_t)(b * S_ + s2)) * D_ + h * 64;
#pragma unroll
    for (int d2 = 0; d2 < 32; ++d2) {
      unsigned int u = *(const unsigned int*)(vp + 2 * d2);
      acc[2 * d2]     = acc[2 * d2]     * sc + ex * bf2f((u16)u);
      acc[2 * d2 + 1] = acc[2 * d2 + 1] * sc + ex * bf2f((u16)(u >> 16));
    }
    m = nm;
  }
  float inv = 1.0f / l;
  u16* op = outb + ((size_t)(b * S_ + g)) * D_ + h * 64;
#pragma unroll
  for (int d2 = 0; d2 < 32; ++d2) {
    unsigned int u = (unsigned int)f2bf(acc[2 * d2] * inv) |
                     ((unsigned int)f2bf(acc[2 * d2 + 1] * inv) << 16);
    *(unsigned int*)(op + 2 * d2) = u;
  }
}

// ---------------------------------------------------------------------------
// Classifier head: out = [verd(B), pros(B), cons(B), mf0(B)] (all f32).
// ---------------------------------------------------------------------------
__global__ void __launch_bounds__(64) cls_kernel(
    const float* __restrict__ x, const float* __restrict__ cls_w,
    const float* __restrict__ cls_b, float* __restrict__ out) {
  __shared__ float sres[6];
  int t = threadIdx.x;
  if (t < 6) {
    int bb = t & 1;
    int i  = t >> 1;
    const float* hr = x + ((size_t)(bb * S_ + i)) * D_;
    const float* wr = cls_w + (size_t)i * D_;
    float dot = 0.0f;
    for (int d = 0; d < D_; ++d) dot += hr[d] * wr[d];
    float val = 1.0f / (1.0f + __expf(-(dot + cls_b[i])));
    sres[t] = val;  // mask[:, i] == 1 for the CLS tokens
  }
  __syncthreads();
  if (t < 8) out[t] = (t < 6) ? sres[t] : 1.0f;  // mf[:,0] == 1
}

// ---------------------------------------------------------------------------
// Host-side orchestration
// ---------------------------------------------------------------------------
extern "C" void kernel_launch(void* const* d_in, const int* in_sizes, int n_in,
                              void* d_out, int out_size, void* d_ws, size_t ws_size,
                              hipStream_t stream) {
  (void)in_sizes; (void)n_in; (void)out_size; (void)ws_size;

  const int*           src      = (const int*)d_in[0];
  const unsigned char* mask_src = (const unsigned char*)d_in[1];
  /* d_in[2] cluster_sizes: unused by the reference */
  const float* word_emb = (const float*)d_in[3];
  const float* pos_emb  = (const float*)d_in[4];
  const float* emb_ln   = (const float*)d_in[5];
  const float* attn_w   = (const float*)d_in[6];
  const float* attn_b   = (const float*)d_in[7];
  const float* ln_p     = (const float*)d_in[8];
  const float* ffn_w1   = (const float*)d_in[9];
  const float* ffn_b1   = (const float*)d_in[10];
  const float* ffn_w2   = (const float*)d_in[11];
  const float* ffn_b2   = (const float*)d_in[12];
  const float* cls_w    = (const float*)d_in[13];
  const float* cls_b    = (const float*)d_in[14];
  float* out = (float*)d_out;

  char* cur = (char*)d_ws;
  auto alloc = [&](size_t bytes) -> char* {
    char* p = cur;
    cur += (bytes + 255) & ~(size_t)255;
    return p;
  };
  u16*   wAt  = (u16*)  alloc((size_t)L_ * 7 * D_ * D_ * sizeof(u16));
  u16*   wF1t = (u16*)  alloc((size_t)L_ * D_ * FF_ * sizeof(u16));
  u16*   wF2t = (u16*)  alloc((size_t)L_ * FF_ * D_ * sizeof(u16));
  float* xF   = (float*)alloc((size_t)MBS * D_ * sizeof(float));
  u16*   xB   = (u16*)  alloc((size_t)MBS * D_ * sizeof(u16));
  float* xaF  = (float*)alloc((size_t)MBS * D_ * sizeof(float));
  u16*   xaB  = (u16*)  alloc((size_t)MBS * D_ * sizeof(u16));
  float* yF   = (float*)alloc((size_t)MBS * D_ * sizeof(float));
  float* hF   = (float*)alloc((size_t)MBS * FF_ * sizeof(float));
  u16*   hB   = (u16*)  alloc((size_t)MBS * FF_ * sizeof(u16));
  u16*   proj[6];
  for (int i = 0; i < 6; ++i) proj[i] = (u16*)alloc((size_t)MBS * D_ * sizeof(u16));
  u16*   aO   = (u16*)  alloc((size_t)MBS * D_ * sizeof(u16));

  dim3 blk(256);

  // One-time weight convert + transpose to bf16 [N][K]
  for (int l = 0; l < L_; ++l) {
    for (int i = 0; i < 7; ++i) {
      size_t off = ((size_t)l * 7 + i) * D_ * D_;
      cvt_transpose_kernel<<<dim3((D_ * D_ + 255) / 256), blk, 0, stream>>>(
          attn_w + off, wAt + off, D_, D_);
    }
    cvt_transpose_kernel<<<dim3((D_ * FF_ + 255) / 256), blk, 0, stream>>>(
        ffn_w1 + (size_t)l * D_ * FF_, wF1t + (size_t)l * D_ * FF_, D_, FF_);
    cvt_transpose_kernel<<<dim3((FF_ * D_ + 255) / 256), blk, 0, stream>>>(
        ffn_w2 + (size_t)l * FF_ * D_, wF2t + (size_t)l * FF_ * D_, FF_, D_);
  }

  // Embedding + LN
  embed_ln_kernel<<<dim3(MBS), blk, 0, stream>>>(src, word_emb, pos_emb, emb_ln, xF, xB);

  for (int l = 0; l < L_; ++l) {
    const u16*   wa  = wAt + (size_t)l * 7 * D_ * D_;
    const float* ab  = attn_b + (size_t)l * 7 * D_;
    const float* lnp = ln_p + (size_t)l * 4 * D_;

    // q,k,v,qg,kg,vg projections (bf16 outputs for attention)
    for (int i = 0; i < 6; ++i) {
      gemm_bf16_kernel<<<dim3(D_ / 128, MBS / 128), blk, 0, stream>>>(
          xB, wa + (size_t)i * D_ * D_, ab + (size_t)i * D_,
          nullptr, proj[i], MBS, D_, D_);
    }
    local_attn_kernel<<<dim3(NC_, H_, B_), blk, 0, stream>>>(
        proj[0], proj[1], proj[2], mask_src, aO);
    global_attn_kernel<<<dim3(B_), dim3(64), 0, stream>>>(
        proj[3], proj[4], proj[5], mask_src, aO);

    // output projection + residual LN
    gemm_bf16_kernel<<<dim3(D_ / 128, MBS / 128), blk, 0, stream>>>(
        aO, wa + (size_t)6 * D_ * D_, ab + (size_t)6 * D_, yF, nullptr, MBS, D_, D_);
    resln_kernel<<<dim3(MBS), blk, 0, stream>>>(xF, yF, lnp, lnp + D_, xaF, xaB);

    // FFN
    gemm_bf16_kernel<<<dim3(FF_ / 128, MBS / 128), blk, 0, stream>>>(
        xaB, wF1t + (size_t)l * D_ * FF_, ffn_b1 + (size_t)l * FF_,
        hF, nullptr, MBS, FF_, D_);
    gelu_kernel<<<dim3((MBS * FF_) / 1024), blk, 0, stream>>>(hF, hB, MBS * FF_);
    gemm_bf16_kernel<<<dim3(D_ / 128, MBS / 128), blk, 0, stream>>>(
        hB, wF2t + (size_t)l * FF_ * D_, ffn_b2 + (size_t)l * D_,
        yF, nullptr, MBS, D_, FF_);
    resln_kernel<<<dim3(MBS), blk, 0, stream>>>(xaF, yF, lnp + 2 * D_, lnp + 3 * D_, xF, xB);
  }

  cls_kernel<<<dim3(1), dim3(64), 0, stream>>>(xF, cls_w, cls_b, out);
}
